// ASG_49813030699390
// MI455X (gfx1250) — compile-verified, matrix operand order checked by reading, not run on hardware
//
#include <hip/hip_runtime.h>
#include <hip/hip_bf16.h>

#define TT 512
#define BB 32
#define VV 512
#define LL 128
#define NEGF (-1.0e9f)

#define PSTR 520        // p-buffer row stride in halves (512 + 8 pad, 16B aligned)
#define ASTR 516        // alpha row stride in floats   (512 + 4 pad, 16B aligned)
#define NBATCH_BLK 16   // batches per block (one 16-wide N tile)

typedef __attribute__((ext_vector_type(16))) _Float16 v16h;
typedef __attribute__((ext_vector_type(8)))  _Float16 v8h;
typedef __attribute__((ext_vector_type(8)))  float    v8f;
typedef __attribute__((ext_vector_type(4)))  float    v4f;

union V16H { v16h v; v8h h[2]; };

// ---------------- kernel 1: E = exp(transition) as fp16 (L2-resident, 512KB) ----------------
__global__ void asg_prep_exp(const float* __restrict__ trans, _Float16* __restrict__ Eh) {
  int i = blockIdx.x * blockDim.x + threadIdx.x;
  Eh[i] = (_Float16)__expf(trans[i]);
}

// ---------------- FCC helpers ----------------
// P phase: wave w (< NBATCH_BLK) owns local batch w. Reads alpha row, computes
// per-batch max m, p = fp16(exp(alpha - m)) into pbuf, and (when t == len-1)
// the logsumexp score fcc[b].
__device__ __forceinline__ void fcc_pphase(int t, int wave, int lane, int bbase,
                                           _Float16* pbuf, float* alpha, float* mb,
                                           const int* __restrict__ in_len,
                                           float* __restrict__ fcc) {
  if (wave >= NBATCH_BLK) return;
  const float* arow = alpha + wave * ASTR + lane * 16;
  float v[16];
  #pragma unroll
  for (int i = 0; i < 4; ++i) {
    v4f x = *(const v4f*)(arow + i * 4);
    #pragma unroll
    for (int j = 0; j < 4; ++j) v[i * 4 + j] = x[j];
  }
  float m = v[0];
  #pragma unroll
  for (int i = 1; i < 16; ++i) m = fmaxf(m, v[i]);
  #pragma unroll
  for (int off = 16; off >= 1; off >>= 1) m = fmaxf(m, __shfl_xor(m, off));

  float s = 0.0f;
  v8h p0, p1;
  #pragma unroll
  for (int i = 0; i < 8; ++i) {
    float e = __expf(v[i] - m);
    s += e;
    p0[i] = (_Float16)e;
  }
  #pragma unroll
  for (int i = 0; i < 8; ++i) {
    float e = __expf(v[8 + i] - m);
    s += e;
    p1[i] = (_Float16)e;
  }
  #pragma unroll
  for (int off = 16; off >= 1; off >>= 1) s += __shfl_xor(s, off);

  _Float16* prow = pbuf + wave * PSTR + lane * 16;
  *(v8h*)(prow)     = p0;
  *(v8h*)(prow + 8) = p1;

  if (lane == 0) {
    mb[wave] = m;
    int b = bbase + wave;
    if (t == in_len[b] - 1) fcc[b] = m + __logf(s);
  }
}

// ---------------- kernel 2: FCC scan. grid = 2 blocks x 1024 threads (32 waves) ------------
// Per step: D[512x16] = E[512x512](f16,global/L2) x P[512x16](f16,LDS) via
// v_wmma_f32_16x16x32_f16; alpha_new = lp_t + m_b + log(D).
__global__ __launch_bounds__(1024) void asg_fcc(const float* __restrict__ lp,
                                                const _Float16* __restrict__ Eh,
                                                const int* __restrict__ in_len,
                                                float* __restrict__ fcc) {
  __shared__ __align__(16) _Float16 pbuf[NBATCH_BLK * PSTR];
  __shared__ __align__(16) float alpha[NBATCH_BLK * ASTR];
  __shared__ float mb[NBATCH_BLK];

  const int tid   = threadIdx.x;
  const int wave  = tid >> 5;
  const int lane  = tid & 31;
  const int bbase = blockIdx.x * NBATCH_BLK;

  // ---- t = 0: alpha = log_probs[0] ----
  if (wave < NBATCH_BLK) {
    const float* src = lp + (size_t)(bbase + wave) * VV;
    float* dst = alpha + wave * ASTR;
    #pragma unroll
    for (int i = 0; i < 4; ++i)
      *(v4f*)(dst + lane * 16 + i * 4) = *(const v4f*)(src + lane * 16 + i * 4);
  }
  __syncthreads();
  fcc_pphase(0, wave, lane, bbase, pbuf, alpha, mb, in_len, fcc);
  __syncthreads();

  const int m0  = wave * 16;      // output row tile (v_new)
  const int row = lane & 15;      // A: matrix row / B: batch column (N)
  const int hi  = lane >> 4;      // K-half select per WMMA lane layout
  const _Float16* Abase = Eh + (size_t)(m0 + row) * VV + hi * 8;
  const _Float16* Bbase = pbuf + row * PSTR + hi * 8;

  for (int t = 1; t < TT; ++t) {
    // Prefetch next timestep's emission slab into cache (global_prefetch_b8).
    if (t + 1 < TT)
      __builtin_prefetch(lp + ((size_t)(t + 1) * BB + bbase) * VV + tid * 8, 0, 0);

    v8f acc = {};
    #pragma unroll 4
    for (int k0 = 0; k0 < VV; k0 += 32) {
      V16H a, bm;
      a.h[0]  = *(const v8h*)(Abase + k0);        // E rows, K+0..7 / K+8..15
      a.h[1]  = *(const v8h*)(Abase + k0 + 16);   // K+16..23 / K+24..31
      bm.h[0] = *(const v8h*)(Bbase + k0);        // p columns, mirrored layout (LDS)
      bm.h[1] = *(const v8h*)(Bbase + k0 + 16);
      acc = __builtin_amdgcn_wmma_f32_16x16x32_f16(false, a.v, false, bm.v,
                                                   (short)0, acc, false, false);
    }

    // Epilogue: alpha_new[v, b] = lp[t,b,v] + m_b + log(acc)
    const int bloc = lane & 15;
    const int dm   = hi * 8;
    const float mbv = mb[bloc];
    const float* lprow = lp + ((size_t)t * BB + bbase + bloc) * VV + m0 + dm;
    v4f l0 = *(const v4f*)(lprow);
    v4f l1 = *(const v4f*)(lprow + 4);
    float* adst = alpha + bloc * ASTR + m0 + dm;
    #pragma unroll
    for (int r = 0; r < 8; ++r) {
      float lpv = (r < 4) ? l0[r] : l1[r - 4];
      adst[r] = lpv + mbv + __logf(acc[r]);
    }
    __syncthreads();
    fcc_pphase(t, wave, lane, bbase, pbuf, alpha, mb, in_len, fcc);
    __syncthreads();
  }
}

// ---------------- kernel 3: FAC banded DP + final combine. grid = 32 x 128 ----------------
__global__ __launch_bounds__(LL) void asg_fac(const float* __restrict__ lp,
                                              const float* __restrict__ trans,
                                              const int* __restrict__ targets,
                                              const int* __restrict__ in_len,
                                              const int* __restrict__ tgt_len,
                                              const float* __restrict__ fcc,
                                              float* __restrict__ out) {
  const int b = blockIdx.x;
  const int l = threadIdx.x;
  __shared__ float sbeta[LL + 1];
  __shared__ float sfac;

  const int tgt = targets[b * LL + l];
  const int tgp = (l > 0) ? targets[b * LL + l - 1] : 0;
  const float ts = trans[tgt * VV + tgt];                      // self transition
  const float tp = (l > 0) ? trans[tgt * VV + tgp] : NEGF;     // prev->cur transition
  const int il = in_len[b] - 1;
  const int tl = tgt_len[b] - 1;

  float beta = (l == 0) ? lp[(size_t)b * VV + tgt] : NEGF;     // beta0
  if (l == 0) sbeta[0] = NEGF;
  sbeta[l + 1] = beta;
  if (il == 0 && l == tl) sfac = beta;

  for (int t = 1; t < TT; ++t) {
    __syncthreads();
    float left = sbeta[l];                                     // beta[l-1]
    float stay = beta + ts;
    float move = left + tp;
    float mx = fmaxf(stay, move);
    float mn = fminf(stay, move);
    float lse = mx + log1pf(__expf(mn - mx));                  // logaddexp
    beta = lp[((size_t)t * BB + b) * VV + tgt] + lse;
    __syncthreads();
    sbeta[l + 1] = beta;
    if (t == il && l == tl) sfac = beta;
  }
  __syncthreads();
  if (l == 0) out[b] = fcc[b] - sfac;
}

// ---------------- host ----------------
extern "C" void kernel_launch(void* const* d_in, const int* in_sizes, int n_in,
                              void* d_out, int out_size, void* d_ws, size_t ws_size,
                              hipStream_t stream) {
  const float* lp      = (const float*)d_in[0];   // [T,B,V] f32
  const float* trans   = (const float*)d_in[1];   // [V,V]   f32
  const int*   targets = (const int*)d_in[2];     // [B,L]   i32
  const int*   in_len  = (const int*)d_in[3];     // [B]     i32
  const int*   tgt_len = (const int*)d_in[4];     // [B]     i32
  float* out = (float*)d_out;                     // [B]     f32

  // workspace: fp16 exp(transition) (512KB) + fcc[32]
  _Float16* Eh  = (_Float16*)d_ws;
  float*    fcc = (float*)((char*)d_ws + (size_t)VV * VV * sizeof(_Float16));

  asg_prep_exp<<<(VV * VV) / 256, 256, 0, stream>>>(trans, Eh);
  asg_fcc<<<BB / NBATCH_BLK, 1024, 0, stream>>>(lp, Eh, in_len, fcc);
  asg_fac<<<BB, LL, 0, stream>>>(lp, trans, targets, in_len, tgt_len, fcc, out);
}
